// G_LSTM_20761871908970
// MI455X (gfx1250) — matrix operand order repaired and checked by reading, not compiled, not run on hardware
//
#include <hip/hip_runtime.h>

typedef _Float16 f16;
typedef __attribute__((ext_vector_type(16))) _Float16 v16h;
typedef __attribute__((ext_vector_type(8)))  _Float16 v8h;
typedef __attribute__((ext_vector_type(8)))  float    v8f;

// global (address_space(1)) pointer views — force global_load_* / global_store_*
typedef const __attribute__((address_space(1))) v8h*   g_v8h;
typedef const __attribute__((address_space(1))) float* g_f32;
typedef       __attribute__((address_space(1))) float* g_f32w;
typedef       __attribute__((address_space(1))) f16*   g_f16w;

#define NB   32768           // batch rows
#define ED   512             // feature dim
#define KCAT 1536            // concat K = |x|+|h|+|c|
#define BE   ((size_t)NB * ED)

__device__ __forceinline__ float sigmoidf_(float x) {
    return 1.0f / (1.0f + __expf(-x));
}

__device__ __forceinline__ v16h cat16(v8h lo, v8h hi) {
    return __builtin_shufflevector(lo, hi, 0,1,2,3,4,5,6,7,8,9,10,11,12,13,14,15);
}

// A-fragment: per-lane chunks at +0..7 and +16..23 f16 (ISA 7.12.2 16-bit A)
__device__ __forceinline__ v16h ldA(const f16* p) {
    g_v8h q = (g_v8h)(uintptr_t)p;
    return cat16(q[0], q[2]);
}
// B-fragment: per-lane chunks at +0..7 and +16..23 f16 along K (matches A)
__device__ __forceinline__ v16h ldB(const f16* p) {
    g_v8h q = (g_v8h)(uintptr_t)p;
    return cat16(q[0], q[2]);
}

// ---------------- pack / convert kernels ----------------

__global__ __launch_bounds__(256) void pack_acts(
    const float* __restrict__ x, const float* __restrict__ h,
    const float* __restrict__ c, f16* __restrict__ XHC) {
    size_t i = (size_t)blockIdx.x * blockDim.x + threadIdx.x;
    if (i >= BE) return;
    size_t row = i / ED, col = i % ED;
    g_f16w dst = (g_f16w)(uintptr_t)(XHC + row * KCAT);
    dst[col        ] = (f16)((g_f32)(uintptr_t)x)[i];
    dst[col + ED   ] = (f16)((g_f32)(uintptr_t)h)[i];
    dst[col + 2*ED ] = (f16)((g_f32)(uintptr_t)c)[i];
}

__global__ __launch_bounds__(256) void pack_weights(
    const float* __restrict__ Wx, const float* __restrict__ Wh,
    const float* __restrict__ Wc, f16* __restrict__ Wcat,
    f16* __restrict__ Wc2) {
    size_t i = (size_t)blockIdx.x * blockDim.x + threadIdx.x;  // over 4*512*512
    if (i >= (size_t)4 * ED * ED) return;
    int    g  = (int)(i / ((size_t)ED * ED));
    size_t fe = i % ((size_t)ED * ED);
    size_t e  = fe % ED;
    g_f16w dst = (g_f16w)(uintptr_t)(Wcat + (i / ED) * KCAT);  // (g*512 + f) row
    dst[e        ] = (f16)Wx[i];
    dst[e + ED   ] = (f16)Wh[i];
    dst[e + 2*ED ] = (g < 2) ? (f16)Wc[(size_t)g * ED * ED + fe] : (f16)0.0f;
    if (g == 2) ((g_f16w)(uintptr_t)Wc2)[fe] = (f16)Wc[(size_t)2 * ED * ED + fe];
}

__global__ __launch_bounds__(256) void pack_bias(
    const float* __restrict__ bx, const float* __restrict__ bh,
    const float* __restrict__ bc, float* __restrict__ biasc) {
    int i = blockIdx.x * blockDim.x + threadIdx.x;   // 4*512
    if (i >= 4 * ED) return;
    int g = i / ED, f = i % ED;
    float v = bx[i] + bh[i];
    if (g < 2)  v += bc[i];            // peephole bias for i,f
    if (g == 2) v += bc[2 * ED + f];   // fold bc[2] into o-partial
    biasc[i] = v;
}

// ---------------- stage 1: fused 4-gate GEMM + cell update ----------------
// block tile 128(M) x 64(N), 8 waves of 32x32, all 4 gates per block.

__global__ __launch_bounds__(256) void lstm_gates_gemm(
    const f16*   __restrict__ XHC,    // [NB, KCAT]
    const f16*   __restrict__ Wcat,   // [4, ED, KCAT]
    const float* __restrict__ biasc,  // [4, ED]
    const float* __restrict__ c_in,   // [NB, ED] fp32
    float* __restrict__ out_c,        // next_c fp32 (d_out + BE)
    f16*   __restrict__ c_bf,         // next_c f16 (stage-2 A matrix)
    float* __restrict__ o_part) {     // gx2+gh2+bias (added to stage-2 GEMM result)
    const int lane = threadIdx.x & 31;
    const int wave = threadIdx.x >> 5;
    const int wm = wave >> 1, wn = wave & 1;
    const int mBase = blockIdx.y * 128 + wm * 32;
    const int nBase = blockIdx.x * 64  + wn * 32;
    const int l15 = lane & 15, lhi = lane >> 4;

    v8f acc[4][2][2];
    #pragma unroll
    for (int g = 0; g < 4; ++g)
        #pragma unroll
        for (int ni = 0; ni < 2; ++ni) {
            float bz = ((g_f32)(uintptr_t)biasc)[g * ED + nBase + ni * 16 + l15];
            #pragma unroll
            for (int mi = 0; mi < 2; ++mi)
                #pragma unroll
                for (int r = 0; r < 8; ++r) acc[g][mi][ni][r] = bz;
        }

    // per-lane fragment pointers (ISA 7.12.2 wave32 layouts)
    const f16* aptr[2];
    #pragma unroll
    for (int mi = 0; mi < 2; ++mi)
        aptr[mi] = XHC + (size_t)(mBase + mi * 16 + l15) * KCAT + lhi * 8;
    const f16* bptr[2];
    #pragma unroll
    for (int ni = 0; ni < 2; ++ni)
        bptr[ni] = Wcat + (size_t)(nBase + ni * 16 + l15) * KCAT + lhi * 8;

    for (int k0 = 0; k0 < KCAT; k0 += 32) {
        v16h a[2];
        #pragma unroll
        for (int mi = 0; mi < 2; ++mi) {
            a[mi] = ldA(aptr[mi]);
            aptr[mi] += 32;
        }
        #pragma unroll
        for (int g = 0; g < 4; ++g) {
            const size_t gOff = (size_t)g * ED * KCAT;
            #pragma unroll
            for (int ni = 0; ni < 2; ++ni) {
                v16h bf = ldB(bptr[ni] + gOff);
                #pragma unroll
                for (int mi = 0; mi < 2; ++mi)
                    acc[g][mi][ni] = __builtin_amdgcn_wmma_f32_16x16x32_f16(
                        false, a[mi], false, bf, (short)0, acc[g][mi][ni],
                        false, false);
            }
        }
        bptr[0] += 32;
        bptr[1] += 32;
    }

    // fused gate epilogue: next_c = sigmoid(f)*c + sigmoid(i)*tanh(g)
    #pragma unroll
    for (int mi = 0; mi < 2; ++mi)
        #pragma unroll
        for (int ni = 0; ni < 2; ++ni)
            #pragma unroll
            for (int r = 0; r < 8; ++r) {
                int m = mBase + mi * 16 + r + lhi * 8;
                int n = nBase + ni * 16 + l15;
                size_t idx = (size_t)m * ED + n;
                float iv = sigmoidf_(acc[0][mi][ni][r]);
                float fv = sigmoidf_(acc[1][mi][ni][r]);
                float gv = tanhf(acc[3][mi][ni][r]);
                float nc = fv * ((g_f32)(uintptr_t)c_in)[idx] + iv * gv;
                ((g_f32w)(uintptr_t)out_c)[idx]  = nc;
                ((g_f16w)(uintptr_t)c_bf)[idx]   = (f16)nc;
                ((g_f32w)(uintptr_t)o_part)[idx] = acc[2][mi][ni][r];
            }
}

// ---------------- stage 2: o-gate peephole GEMM + output ----------------

__global__ __launch_bounds__(256) void lstm_out_gemm(
    const f16*   __restrict__ c_bf,    // [NB, ED]
    const f16*   __restrict__ Wc2,     // [ED, ED]
    const float* __restrict__ o_part,  // [NB, ED]
    const float* __restrict__ next_c,  // fp32 (d_out + BE)
    float* __restrict__ out_h) {
    const int lane = threadIdx.x & 31;
    const int wave = threadIdx.x >> 5;
    const int wm = wave >> 1, wn = wave & 1;
    const int mBase = blockIdx.y * 128 + wm * 32;
    const int nBase = blockIdx.x * 64  + wn * 32;
    const int l15 = lane & 15, lhi = lane >> 4;

    v8f acc[2][2] = {};
    const f16* aptr[2];
    #pragma unroll
    for (int mi = 0; mi < 2; ++mi)
        aptr[mi] = c_bf + (size_t)(mBase + mi * 16 + l15) * ED + lhi * 8;
    const f16* bptr[2];
    #pragma unroll
    for (int ni = 0; ni < 2; ++ni)
        bptr[ni] = Wc2 + (size_t)(nBase + ni * 16 + l15) * ED + lhi * 8;

    for (int k0 = 0; k0 < ED; k0 += 32) {
        v16h a[2], b[2];
        #pragma unroll
        for (int mi = 0; mi < 2; ++mi) {
            a[mi] = ldA(aptr[mi]);
            aptr[mi] += 32;
        }
        #pragma unroll
        for (int ni = 0; ni < 2; ++ni) {
            b[ni] = ldB(bptr[ni]);
            bptr[ni] += 32;
        }
        #pragma unroll
        for (int mi = 0; mi < 2; ++mi)
            #pragma unroll
            for (int ni = 0; ni < 2; ++ni)
                acc[mi][ni] = __builtin_amdgcn_wmma_f32_16x16x32_f16(
                    false, a[mi], false, b[ni], (short)0, acc[mi][ni],
                    false, false);
    }

    #pragma unroll
    for (int mi = 0; mi < 2; ++mi)
        #pragma unroll
        for (int ni = 0; ni < 2; ++ni)
            #pragma unroll
            for (int r = 0; r < 8; ++r) {
                int m = mBase + mi * 16 + r + lhi * 8;
                int n = nBase + ni * 16 + l15;
                size_t idx = (size_t)m * ED + n;
                float o = sigmoidf_(acc[mi][ni][r] + ((g_f32)(uintptr_t)o_part)[idx]);
                ((g_f32w)(uintptr_t)out_h)[idx] =
                    o * ((g_f32)(uintptr_t)next_c)[idx];
            }
}

// ---------------- launcher ----------------

extern "C" void kernel_launch(void* const* d_in, const int* in_sizes, int n_in,
                              void* d_out, int out_size, void* d_ws, size_t ws_size,
                              hipStream_t stream) {
    const float* x  = (const float*)d_in[0];
    const float* h  = (const float*)d_in[1];
    const float* c  = (const float*)d_in[2];
    const float* Wx = (const float*)d_in[3];
    const float* bx = (const float*)d_in[4];
    const float* Wh = (const float*)d_in[5];
    const float* bh = (const float*)d_in[6];
    const float* Wc = (const float*)d_in[7];
    const float* bc = (const float*)d_in[8];
    float* out = (float*)d_out;              // [next_h (BE) | next_c (BE)]

    // workspace carve-up (256B aligned)
    size_t off = 0;
    auto carve = [&](size_t bytes) {
        size_t cur = off;
        off = (off + bytes + 255) & ~(size_t)255;
        return (char*)d_ws + cur;
    };
    f16*   XHC    = (f16*)  carve(BE * 3 * sizeof(f16));            // [NB,1536]
    f16*   Wcat   = (f16*)  carve((size_t)4 * ED * KCAT * sizeof(f16));
    f16*   Wc2    = (f16*)  carve((size_t)ED * ED * sizeof(f16));
    float* biasc  = (float*)carve((size_t)4 * ED * sizeof(float));
    float* o_part = (float*)carve(BE * sizeof(float));
    f16*   c_bf   = (f16*)  carve(BE * sizeof(f16));

    float* next_c = out + BE;
    float* next_h = out;

    pack_acts<<<(unsigned)((BE + 255) / 256), 256, 0, stream>>>(x, h, c, XHC);
    pack_weights<<<(unsigned)(((size_t)4 * ED * ED + 255) / 256), 256, 0, stream>>>(
        Wx, Wh, Wc, Wcat, Wc2);
    pack_bias<<<8, 256, 0, stream>>>(bx, bh, bc, biasc);

    dim3 grid1(ED / 64, NB / 128);   // 8 x 256
    lstm_gates_gemm<<<grid1, 256, 0, stream>>>(XHC, Wcat, biasc, c, next_c,
                                               c_bf, o_part);
    lstm_out_gemm<<<grid1, 256, 0, stream>>>(c_bf, Wc2, o_part, next_c, next_h);
}
